// ContinuousThoughtMachineAttn_7327214207638
// MI455X (gfx1250) — matrix operand order repaired and checked by reading, not compiled, not run on hardware
//
#include <hip/hip_runtime.h>
#include <hip/hip_bf16.h>

// ---------------------------------------------------------------------------
// Continuous Thought Machine on MI455X (gfx1250, wave32, WMMA bf16 + TDM)
// ---------------------------------------------------------------------------

typedef __bf16 bf16_t;
typedef bf16_t v16bf __attribute__((ext_vector_type(16)));
typedef float  v8f   __attribute__((ext_vector_type(8)));
typedef unsigned int u32x4 __attribute__((ext_vector_type(4)));
typedef int i32x8 __attribute__((ext_vector_type(8)));
typedef int i32x4 __attribute__((ext_vector_type(4)));

#define DIV_UP(a, b) (((a) + (b) - 1) / (b))

// Probe for the Tensor Data Mover builtin (this toolchain: 6-arg form,
// (u32x4 g0, i32x8 g1, i32x4 g2, i32x4 g3, i32x8 g4, i32 cpol))
#if defined(__AMDGCN__) && defined(__has_builtin)
#if __has_builtin(__builtin_amdgcn_tensor_load_to_lds) && \
    __has_builtin(__builtin_amdgcn_s_wait_tensorcnt)
#define CTM_HAVE_TDM 1
#endif
#endif

// Problem constants
#define CB     64      // batch
#define CS     196     // sequence
#define CFEAT  512
#define CDM    1024    // d_model
#define CDI    512     // d_input
#define CH     8       // cross heads (dh = 64)
#define CSH    8       // sync heads  (dh_sync = 128)
#define CIT    50      // ticks
#define CMEM   25      // memory length
#define CMEMP  32      // padded memory (K for WMMA)
#define COUT   1000
#define COUTP  1008    // padded to multiple of 16

#define PRED_ELEMS (CB * COUT * CIT)        // 3,200,000
#define CERT_OFF   PRED_ELEMS
#define CERT_ELEMS (CB * 2 * CIT)           // 6,400
#define SYNC_OFF   (CERT_OFF + CERT_ELEMS)  // 3,206,400

// ---------------------------------------------------------------------------
// TDM helper: async DMA a 2D bf16 tile (rows x row_elems, row stride
// src_stride_elems) from global memory into LDS.  Descriptor built per the
// CDNA5 ISA D# layout (CDNA5 ISA).  Caller must issue
// from a single wave and follow with tdm_wait() + __syncthreads().
// ---------------------------------------------------------------------------
#ifdef CTM_HAVE_TDM
__device__ __forceinline__ void tdm_load_2d_bf16(const bf16_t* gsrc, void* lds_dst,
                                                 unsigned rows, unsigned row_elems,
                                                 unsigned src_stride_elems) {
  unsigned long long ga = (unsigned long long)(size_t)gsrc;
  unsigned ldsa = (unsigned)(size_t)lds_dst;  // low 32 bits = LDS byte offset
  u32x4 g0;
  g0[0] = 1u;                                        // count=1, user mode
  g0[1] = ldsa;                                      // lds_addr
  g0[2] = (unsigned)ga;                              // global_addr[31:0]
  g0[3] = ((unsigned)(ga >> 32) & 0x01FFFFFFu) | 0x80000000u;  // [56:32] | type=2
  i32x8 g1;
  g1[0] = (int)(1u << 16);                           // data_size = 1 (2 bytes)
  g1[1] = (int)((row_elems & 0xFFFFu) << 16);        // tensor_dim0[15:0]
  g1[2] = (int)(((row_elems >> 16) & 0xFFFFu) | ((rows & 0xFFFFu) << 16));
  g1[3] = (int)(((rows >> 16) & 0xFFFFu) | ((row_elems & 0xFFFFu) << 16));  // tile_dim0
  g1[4] = (int)(rows & 0xFFFFu);                     // tile_dim1 (tile_dim2 = 0)
  g1[5] = (int)src_stride_elems;                     // tensor_dim0_stride[31:0]
  g1[6] = 0;                                         // stride[47:32], dim1_stride lo
  g1[7] = 0;
  i32x4 gz4 = {0, 0, 0, 0};
  i32x8 gz8 = {0, 0, 0, 0, 0, 0, 0, 0};
  __builtin_amdgcn_tensor_load_to_lds(g0, g1, gz4, gz4, gz8, 0);
}
__device__ __forceinline__ void tdm_wait() {
  __builtin_amdgcn_s_wait_tensorcnt(0);
}
#endif

// ---------------------------------------------------------------------------
// Weight conversion: W (N x K) f32 row-major  ->  Wt (K x Npad) bf16 row-major
// ---------------------------------------------------------------------------
__global__ void k_conv_t(const float* __restrict__ W, bf16_t* __restrict__ out,
                         int N, int K, int Npad) {
  int idx = blockIdx.x * 256 + threadIdx.x;
  if (idx >= K * Npad) return;
  int k = idx / Npad, n = idx % Npad;
  out[idx] = (n < N) ? (bf16_t)W[(size_t)n * K + k] : (bf16_t)0.0f;
}

// ---------------------------------------------------------------------------
// Generic WMMA GEMM:  C(M,N) = A(M,K)f32 * B(K,N)bf16 (+bias)
// 4 waves/block, one 16x16 tile per wave.  All launches have (M/16)%4==0, so
// every block covers one 16-wide N column: the K x 16 bf16 weight slab is
// DMA'd into LDS once via the Tensor Data Mover, then fragments are read from
// LDS.  A and B fragments both use contiguous k order (k = 16*(lane/16) + s).
// ---------------------------------------------------------------------------
__launch_bounds__(128)
__global__ void k_gemm(const float* __restrict__ A, int lda,
                       const bf16_t* __restrict__ B, int ldb,
                       const float* __restrict__ bias, int nbias,
                       float* __restrict__ C, int ldc,
                       int M, int N, int K) {
  __shared__ bf16_t btile[1536 * 16];  // K x 16 slab (max K = 1536), 48 KB
  const int lane = threadIdx.x & 31;
  const int wave = threadIdx.x >> 5;
  const int mt = M >> 4;
  const int tile = blockIdx.x * 4 + wave;
  const int n0 = ((blockIdx.x * 4) / mt) << 4;  // uniform across block
  const int m0 = (tile % mt) << 4;

  // stage B[k=0..K-1][n0..n0+15] into LDS
#ifdef CTM_HAVE_TDM
  if (threadIdx.x < 32) {  // wave 0 issues the DMA (TDM ignores EXEC)
    tdm_load_2d_bf16(B + n0, btile, (unsigned)K, 16u, (unsigned)ldb);
    tdm_wait();
  }
#else
  for (int i = threadIdx.x; i < K * 16; i += 128)
    btile[i] = B[(size_t)(i >> 4) * ldb + n0 + (i & 15)];
#endif
  __syncthreads();

  const int r   = lane & 15;
  const int akb = (lane < 16) ? 0 : 16;  // A: K-base per lane half
  const int bkb = (lane < 16) ? 0 : 16;  // B: K-base per lane half

  v8f acc = {0.f, 0.f, 0.f, 0.f, 0.f, 0.f, 0.f, 0.f};

  for (int k0 = 0; k0 < K; k0 += 32) {
    const float* Ap = A + (size_t)(m0 + r) * lda + k0;
    __builtin_prefetch(Ap + 32, 0, 1);  // global_prefetch_b8 on gfx1250
    v16bf af;
#pragma unroll
    for (int v = 0; v < 8; ++v) {
      int kA = akb + 2 * v;
      af[2 * v]     = (bf16_t)Ap[kA];
      af[2 * v + 1] = (bf16_t)Ap[kA + 1];
    }
    const bf16_t* Bp = btile + (k0 + bkb) * 16 + r;
    v16bf bfr;
#pragma unroll
    for (int v = 0; v < 8; ++v) {
      bfr[2 * v]     = Bp[(2 * v) * 16];
      bfr[2 * v + 1] = Bp[(2 * v + 1) * 16];
    }
    acc = __builtin_amdgcn_wmma_f32_16x16x32_bf16(false, af, false, bfr,
                                                  (short)0, acc, false, false);
  }

  const int col = n0 + r;
  const float bv = (bias != nullptr && col < nbias) ? bias[col] : 0.0f;
  const int rbase = m0 + ((lane < 16) ? 0 : 8);
#pragma unroll
  for (int i = 0; i < 8; ++i)
    C[(size_t)(rbase + i) * ldc + col] = acc[i] + bv;
}

// ---------------------------------------------------------------------------
// In-place LayerNorm over rows of length L (block per row)
// ---------------------------------------------------------------------------
__launch_bounds__(256)
__global__ void k_ln_rows(float* __restrict__ x, const float* __restrict__ s,
                          const float* __restrict__ bb, int L) {
  __shared__ float r1[256], r2[256];
  const int row = blockIdx.x, tid = threadIdx.x;
  float* xp = x + (size_t)row * L;
  float sm = 0.f, sq = 0.f;
  for (int i = tid; i < L; i += 256) { float v = xp[i]; sm += v; sq += v * v; }
  r1[tid] = sm; r2[tid] = sq; __syncthreads();
  for (int o = 128; o >= 1; o >>= 1) {
    if (tid < o) { r1[tid] += r1[tid + o]; r2[tid] += r2[tid + o]; }
    __syncthreads();
  }
  const float mean = r1[0] / L;
  const float var  = r2[0] / L - mean * mean;
  const float rstd = rsqrtf(var + 1e-5f);
  for (int i = tid; i < L; i += 256)
    xp[i] = (xp[i] - mean) * rstd * s[i] + bb[i];
}

// ---------------------------------------------------------------------------
// Fused neuron-sync attention: per (b,h) block.
//   TDM-stage the 32x128 Q and K bf16 tiles into LDS,
//   scores(128x128) = Qc_h^T * Kc_h (K = 32 padded slots) via WMMA,
//   softmax rows + contraction with V, output interleaved d*8+h.
// ---------------------------------------------------------------------------
__launch_bounds__(256)
__global__ void k_sync_attn(const bf16_t* __restrict__ Qc,
                            const bf16_t* __restrict__ Kc,
                            const float* __restrict__ vcur,
                            float* __restrict__ syncv) {
  __shared__ bf16_t qt[CMEMP * 128];   // 8 KB
  __shared__ bf16_t kt[CMEMP * 128];   // 8 KB
  __shared__ bf16_t sc[128 * 130];     // ~33 KB
  __shared__ float Vsh[128];
  const int b = blockIdx.x >> 3;
  const int h = blockIdx.x & 7;
  const int tid = threadIdx.x;
  const int lane = tid & 31;
  const int wave = tid >> 5;
  const size_t base = (size_t)b * (CMEMP * CDM) + (size_t)h * 128;

  if (tid < 128) Vsh[tid] = vcur[(size_t)b * CDM + h * 128 + tid];

  // stage Q/K tiles (32 rows x 128 cols, row stride CDM) into LDS
#ifdef CTM_HAVE_TDM
  if (tid < 32) {  // wave 0 issues both DMAs
    tdm_load_2d_bf16(Qc + base, qt, CMEMP, 128u, CDM);
    tdm_load_2d_bf16(Kc + base, kt, CMEMP, 128u, CDM);
    tdm_wait();
  }
#else
  for (int i = tid; i < CMEMP * 128; i += 256) {
    int t = i >> 7, e = i & 127;
    qt[i] = Qc[base + (size_t)t * CDM + e];
    kt[i] = Kc[base + (size_t)t * CDM + e];
  }
#endif
  __syncthreads();

  const int r   = lane & 15;
  const int akb = (lane < 16) ? 0 : 16;
  const int bkb = (lane < 16) ? 0 : 16;
  const int m0  = wave * 16;

  // A fragment: A[d][t] = qt[t*128 + m0 + d]  (column-major source)
  v16bf af;
  const bf16_t* Ap = qt + m0 + r;
#pragma unroll
  for (int v = 0; v < 8; ++v) {
    int kA = akb + 2 * v;
    af[2 * v]     = Ap[kA * 128];
    af[2 * v + 1] = Ap[(kA + 1) * 128];
  }

  const float scale = 0.0883883476483184f;  // 1/sqrt(128)
  for (int ntile = 0; ntile < 8; ++ntile) {
    const int n0 = ntile * 16;
    const bf16_t* Bp = kt + n0 + r;
    v16bf bfr;
#pragma unroll
    for (int v = 0; v < 8; ++v) {
      bfr[2 * v]     = Bp[(bkb + 2 * v) * 128];
      bfr[2 * v + 1] = Bp[(bkb + 2 * v + 1) * 128];
    }
    v8f acc = {0.f, 0.f, 0.f, 0.f, 0.f, 0.f, 0.f, 0.f};
    acc = __builtin_amdgcn_wmma_f32_16x16x32_bf16(false, af, false, bfr,
                                                  (short)0, acc, false, false);
    const int col = n0 + r;
    const int rb  = m0 + ((lane < 16) ? 0 : 8);
#pragma unroll
    for (int i = 0; i < 8; ++i)
      sc[(rb + i) * 130 + col] = (bf16_t)(acc[i] * scale);
  }
  __syncthreads();

  // softmax over e + contraction with V; wave handles 16 rows
  for (int d = wave * 16; d < wave * 16 + 16; ++d) {
    float x0 = (float)sc[d * 130 + lane];
    float x1 = (float)sc[d * 130 + lane + 32];
    float x2 = (float)sc[d * 130 + lane + 64];
    float x3 = (float)sc[d * 130 + lane + 96];
    float m = fmaxf(fmaxf(x0, x1), fmaxf(x2, x3));
#pragma unroll
    for (int o = 16; o >= 1; o >>= 1) m = fmaxf(m, __shfl_xor(m, o, 32));
    float e0 = expf(x0 - m), e1 = expf(x1 - m), e2 = expf(x2 - m), e3 = expf(x3 - m);
    float den = e0 + e1 + e2 + e3;
    float num = e0 * Vsh[lane] + e1 * Vsh[lane + 32] +
                e2 * Vsh[lane + 64] + e3 * Vsh[lane + 96];
#pragma unroll
    for (int o = 16; o >= 1; o >>= 1) {
      den += __shfl_xor(den, o, 32);
      num += __shfl_xor(num, o, 32);
    }
    if (lane == 0)
      syncv[(size_t)b * CDM + d * CSH + h] = num / den;  // interleave: d*8+h
  }
}

// ---------------------------------------------------------------------------
// Cross attention: per (b,h) block; query length 1, S=196 keys, dh=64.
// ---------------------------------------------------------------------------
__launch_bounds__(256)
__global__ void k_cross_attn(const float* __restrict__ Qh,
                             const float* __restrict__ Kc,
                             const float* __restrict__ Vc,
                             float* __restrict__ oatt) {
  __shared__ float qs[64];
  __shared__ float sc[208];
  __shared__ float red[256];
  const int b = blockIdx.x >> 3, h = blockIdx.x & 7, tid = threadIdx.x;
  if (tid < 64) qs[tid] = Qh[(size_t)b * CDI + h * 64 + tid];
  __syncthreads();

  float s = -1e30f;
  if (tid < CS) {
    const float* kp = Kc + ((size_t)(b * CS + tid)) * CDI + h * 64;
    float acc = 0.f;
    for (int d = 0; d < 64; ++d) acc += qs[d] * kp[d];
    s = acc * 0.125f;  // 1/sqrt(64)
  }
  red[tid] = s; __syncthreads();
  for (int o = 128; o >= 1; o >>= 1) {
    if (tid < o) red[tid] = fmaxf(red[tid], red[tid + o]);
    __syncthreads();
  }
  const float m = red[0]; __syncthreads();

  float e = (tid < CS) ? expf(s - m) : 0.f;
  if (tid < CS) sc[tid] = e;
  red[tid] = e; __syncthreads();
  for (int o = 128; o >= 1; o >>= 1) {
    if (tid < o) red[tid] += red[tid + o];
    __syncthreads();
  }
  const float inv = 1.f / red[0]; __syncthreads();

  if (tid < 64) {
    const float* vp = Vc + ((size_t)b * CS) * CDI + h * 64 + tid;
    float acc = 0.f;
    for (int k = 0; k < CS; ++k) acc += sc[k] * vp[(size_t)k * CDI];
    oatt[(size_t)b * CDI + h * 64 + tid] = acc * inv;
  }
}

// pre = concat(ao (B,512), act (B,1024)) -> (B,1536)
__global__ void k_concat(const float* __restrict__ ao, const float* __restrict__ act,
                         float* __restrict__ pre) {
  int idx = blockIdx.x * 256 + threadIdx.x;
  if (idx >= CB * 1536) return;
  int b = idx / 1536, j = idx % 1536;
  pre[idx] = (j < CDI) ? ao[(size_t)b * CDI + j] : act[(size_t)b * CDM + (j - CDI)];
}

// synapse: GLU(z) -> LayerNorm -> write into sliding trace slot (st layout (b,d,m))
__launch_bounds__(256)
__global__ void k_syn_glu_ln(const float* __restrict__ z, const float* __restrict__ s,
                             const float* __restrict__ bb, float* __restrict__ st,
                             int slot) {
  __shared__ float hsh[CDM];
  __shared__ float r1[256], r2[256];
  const int b = blockIdx.x, tid = threadIdx.x;
  const float* zp = z + (size_t)b * 2048;
  float sm = 0.f, sq = 0.f;
  for (int d = tid; d < CDM; d += 256) {
    float a = zp[d], g = zp[CDM + d];
    float hv = a * (1.f / (1.f + expf(-g)));
    hsh[d] = hv; sm += hv; sq += hv * hv;
  }
  r1[tid] = sm; r2[tid] = sq; __syncthreads();
  for (int o = 128; o >= 1; o >>= 1) {
    if (tid < o) { r1[tid] += r1[tid + o]; r2[tid] += r2[tid + o]; }
    __syncthreads();
  }
  const float mean = r1[0] / CDM;
  const float var  = r2[0] / CDM - mean * mean;
  const float rstd = rsqrtf(var + 1e-5f);
  for (int d = tid; d < CDM; d += 256)
    st[(size_t)b * (CDM * CMEM) + (size_t)d * CMEM + slot] =
        (hsh[d] - mean) * rstd * s[d] + bb[d];
}

// per-neuron deep NLM; thread = (b,d). slot(m) = (tick+1+m)%25.
__launch_bounds__(256)
__global__ void k_nlm(const float* __restrict__ st, const float* __restrict__ w1,
                      const float* __restrict__ b1, const float* __restrict__ w2,
                      const float* __restrict__ b2, float* __restrict__ act, int tick) {
  const int d = blockIdx.x * 256 + threadIdx.x;  // gridDim.x = 4
  const int b = blockIdx.y;
  float h1[32];
#pragma unroll
  for (int j = 0; j < 32; ++j) h1[j] = b1[d * 32 + j];
  const float* stp = st + (size_t)b * (CDM * CMEM) + (size_t)d * CMEM;
  for (int m = 0; m < CMEM; ++m) {
    const int slot = (tick + 1 + m) % CMEM;
    const float sv = stp[slot];
    const float* wp = w1 + (size_t)m * 32 * CDM + d;
#pragma unroll
    for (int j = 0; j < 32; ++j) h1[j] += sv * wp[(size_t)j * CDM];
  }
  float o0 = b2[d * 2 + 0], o1 = b2[d * 2 + 1];
#pragma unroll
  for (int j = 0; j < 16; ++j) {
    float hv = h1[j] * (1.f / (1.f + expf(-h1[16 + j])));
    o0 += hv * w2[(size_t)(j * 2 + 0) * CDM + d];
    o1 += hv * w2[(size_t)(j * 2 + 1) * CDM + d];
  }
  act[(size_t)b * CDM + d] = o0 * (1.f / (1.f + expf(-o1)));
}

// write new Q/K projections into circular bf16 caches at `slot`
__global__ void k_cache_update(const float* __restrict__ qn, const float* __restrict__ kn,
                               bf16_t* __restrict__ Qc, bf16_t* __restrict__ Kc, int slot) {
  int idx = blockIdx.x * 256 + threadIdx.x;
  if (idx >= CB * CDM) return;
  int b = idx >> 10, n = idx & 1023;
  size_t o = (size_t)b * (CMEMP * CDM) + (size_t)slot * CDM + n;
  Qc[o] = (bf16_t)qn[idx];
  Kc[o] = (bf16_t)kn[idx];
}

// predictions + certainty (entropy of softmax) written strided into d_out
__launch_bounds__(256)
__global__ void k_pred_out(const float* __restrict__ pb, float* __restrict__ out, int t) {
  __shared__ float red[256];
  const int b = blockIdx.x, tid = threadIdx.x;
  const float* p = pb + (size_t)b * COUTP;
  float mx = -1e30f;
  for (int i = tid; i < COUT; i += 256) mx = fmaxf(mx, p[i]);
  red[tid] = mx; __syncthreads();
  for (int o = 128; o >= 1; o >>= 1) {
    if (tid < o) red[tid] = fmaxf(red[tid], red[tid + o]);
    __syncthreads();
  }
  mx = red[0]; __syncthreads();
  float se = 0.f;
  for (int i = tid; i < COUT; i += 256) se += expf(p[i] - mx);
  red[tid] = se; __syncthreads();
  for (int o = 128; o >= 1; o >>= 1) {
    if (tid < o) red[tid] += red[tid + o];
    __syncthreads();
  }
  const float logZ = mx + logf(red[0]); __syncthreads();
  float ent = 0.f;
  for (int i = tid; i < COUT; i += 256) {
    float lp = p[i] - logZ;
    ent += expf(lp) * lp;
    out[(size_t)b * (COUT * CIT) + (size_t)i * CIT + t] = p[i];
  }
  red[tid] = ent; __syncthreads();
  for (int o = 128; o >= 1; o >>= 1) {
    if (tid < o) red[tid] += red[tid + o];
    __syncthreads();
  }
  if (tid == 0) {
    const float ne = -red[0] / 6.907755278982137f;  // log(1000)
    float* cert = out + CERT_OFF;
    cert[(size_t)b * (2 * CIT) + t] = ne;
    cert[(size_t)b * (2 * CIT) + CIT + t] = 1.f - ne;
  }
}

// --- init kernels -----------------------------------------------------------
__global__ void k_init_proj(const float* __restrict__ start, const float* __restrict__ Wq,
                            const float* __restrict__ Wk, const float* __restrict__ Wv,
                            float* __restrict__ qkv0) {
  int idx = blockIdx.x * 256 + threadIdx.x;
  if (idx >= 3 * CDM) return;
  int w = idx >> 10, n = idx & 1023;
  const float* W = (w == 0) ? Wq : ((w == 1) ? Wk : Wv);
  const float* wr = W + (size_t)n * CDM;
  float acc = 0.f;
  for (int k = 0; k < CDM; ++k) acc += start[k] * wr[k];
  qkv0[idx] = acc;
}

__global__ void k_fill_cache(const float* __restrict__ qkv0,
                             bf16_t* __restrict__ Qc, bf16_t* __restrict__ Kc) {
  int idx = blockIdx.x * 256 + threadIdx.x;
  if (idx >= CB * CMEMP * CDM) return;
  int n = idx & 1023;
  int slot = (idx >> 10) & 31;
  Qc[idx] = (bf16_t)((slot < CMEM) ? qkv0[n] : 0.f);
  Kc[idx] = (bf16_t)((slot < CMEM) ? qkv0[CDM + n] : 0.f);
}

__global__ void k_fill_act_v(const float* __restrict__ qkv0, const float* __restrict__ start,
                             float* __restrict__ vcur, float* __restrict__ act) {
  int idx = blockIdx.x * 256 + threadIdx.x;
  if (idx >= CB * CDM) return;
  int n = idx & 1023;
  vcur[idx] = qkv0[2 * CDM + n];
  act[idx]  = start[n];
}

__global__ void k_fill_st(const float* __restrict__ trace, float* __restrict__ st) {
  int idx = blockIdx.x * 256 + threadIdx.x;
  if (idx >= CB * CDM * CMEM) return;
  st[idx] = trace[idx % (CDM * CMEM)];  // (d,m) row-major matches st (b,d,m)
}

__global__ void k_copy_sync(const float* __restrict__ syncv, float* __restrict__ out) {
  int idx = blockIdx.x * 256 + threadIdx.x;
  if (idx >= CB * CDM) return;
  out[SYNC_OFF + idx] = syncv[idx];
}

// ---------------------------------------------------------------------------
// Host orchestration
// ---------------------------------------------------------------------------
extern "C" void kernel_launch(void* const* d_in, const int* in_sizes, int n_in,
                              void* d_out, int out_size, void* d_ws, size_t ws_size,
                              hipStream_t stream) {
  (void)in_sizes; (void)n_in; (void)out_size; (void)ws_size;

  const float* x         = (const float*)d_in[0];
  const float* kv_w      = (const float*)d_in[1];
  const float* kv_b      = (const float*)d_in[2];
  const float* ln_kv_s   = (const float*)d_in[3];
  const float* ln_kv_b   = (const float*)d_in[4];
  const float* q_w       = (const float*)d_in[5];
  const float* q_b       = (const float*)d_in[6];
  const float* in_proj_w = (const float*)d_in[7];
  const float* in_proj_b = (const float*)d_in[8];
  const float* out_proj_w= (const float*)d_in[9];
  const float* out_proj_b= (const float*)d_in[10];
  const float* W_q       = (const float*)d_in[11];
  const float* W_k       = (const float*)d_in[12];
  const float* W_v       = (const float*)d_in[13];
  const float* syn_w     = (const float*)d_in[14];
  const float* syn_b     = (const float*)d_in[15];
  const float* syn_ln_s  = (const float*)d_in[16];
  const float* syn_ln_b  = (const float*)d_in[17];
  const float* nlm_w1    = (const float*)d_in[18];
  const float* nlm_b1    = (const float*)d_in[19];
  const float* nlm_w2    = (const float*)d_in[20];
  const float* nlm_b2    = (const float*)d_in[21];
  const float* start_act = (const float*)d_in[22];
  const float* start_tr  = (const float*)d_in[23];
  const float* out_w     = (const float*)d_in[24];
  const float* out_b     = (const float*)d_in[25];
  float* out = (float*)d_out;

  // ---- workspace bump allocator ----
  char* wsb = (char*)d_ws;
  size_t off = 0;
  auto alloc = [&](size_t bytes) -> void* {
    off = (off + 255) & ~(size_t)255;
    void* p = wsb + off;
    off += bytes;
    return p;
  };

  // bf16 transposed weights (K x Npad)
  bf16_t* t_kv   = (bf16_t*)alloc((size_t)512 * 512 * 2);
  bf16_t* t_wq_c = (bf16_t*)alloc((size_t)512 * 512 * 2);
  bf16_t* t_wk_c = (bf16_t*)alloc((size_t)512 * 512 * 2);
  bf16_t* t_wv_c = (bf16_t*)alloc((size_t)512 * 512 * 2);
  bf16_t* t_qw   = (bf16_t*)alloc((size_t)1024 * 512 * 2);
  bf16_t* t_op   = (bf16_t*)alloc((size_t)512 * 512 * 2);
  bf16_t* t_Wq   = (bf16_t*)alloc((size_t)1024 * 1024 * 2);
  bf16_t* t_Wk   = (bf16_t*)alloc((size_t)1024 * 1024 * 2);
  bf16_t* t_Wv   = (bf16_t*)alloc((size_t)1024 * 1024 * 2);
  bf16_t* t_syn  = (bf16_t*)alloc((size_t)1536 * 2048 * 2);
  bf16_t* t_out  = (bf16_t*)alloc((size_t)1024 * COUTP * 2);
  // bf16 circular sync caches (B, 32, 1024)
  bf16_t* Qcache = (bf16_t*)alloc((size_t)CB * CMEMP * CDM * 2);
  bf16_t* Kcache = (bf16_t*)alloc((size_t)CB * CMEMP * CDM * 2);
  // f32 buffers
  float* kvbuf  = (float*)alloc((size_t)CB * CS * CDI * 4);
  float* Kcbuf  = (float*)alloc((size_t)CB * CS * CDI * 4);
  float* Vcbuf  = (float*)alloc((size_t)CB * CS * CDI * 4);
  float* qkv0   = (float*)alloc((size_t)3 * CDM * 4);
  float* syncv  = (float*)alloc((size_t)CB * CDM * 4);
  float* qbuf   = (float*)alloc((size_t)CB * CDI * 4);
  float* qh     = (float*)alloc((size_t)CB * CDI * 4);
  float* oatt   = (float*)alloc((size_t)CB * CDI * 4);
  float* aobuf  = (float*)alloc((size_t)CB * CDI * 4);
  float* pre    = (float*)alloc((size_t)CB * 1536 * 4);
  float* zbuf   = (float*)alloc((size_t)CB * 2048 * 4);
  float* stbuf  = (float*)alloc((size_t)CB * CDM * CMEM * 4);
  float* actbuf = (float*)alloc((size_t)CB * CDM * 4);
  float* qnew   = (float*)alloc((size_t)CB * CDM * 4);
  float* knew   = (float*)alloc((size_t)CB * CDM * 4);
  float* vbuf   = (float*)alloc((size_t)CB * CDM * 4);
  float* predb  = (float*)alloc((size_t)CB * COUTP * 4);

  auto convt = [&](const float* W, bf16_t* o, int N, int K, int Npad) {
    int n = K * Npad;
    k_conv_t<<<DIV_UP(n, 256), 256, 0, stream>>>(W, o, N, K, Npad);
  };
  auto gemm = [&](const float* A, int lda, const bf16_t* Bm, int ldb,
                  const float* bias, int nbias, float* C, int ldc,
                  int M, int N, int K) {
    int tiles = (M / 16) * (N / 16);
    k_gemm<<<DIV_UP(tiles, 4), 128, 0, stream>>>(A, lda, Bm, ldb, bias, nbias,
                                                 C, ldc, M, N, K);
  };

  // ---- phase 0: weight conversion ----
  convt(kv_w, t_kv, 512, 512, 512);
  convt(in_proj_w + 0 * 512 * 512, t_wq_c, 512, 512, 512);
  convt(in_proj_w + 1 * 512 * 512, t_wk_c, 512, 512, 512);
  convt(in_proj_w + 2 * 512 * 512, t_wv_c, 512, 512, 512);
  convt(q_w, t_qw, 512, 1024, 512);
  convt(out_proj_w, t_op, 512, 512, 512);
  convt(W_q, t_Wq, 1024, 1024, 1024);
  convt(W_k, t_Wk, 1024, 1024, 1024);
  convt(W_v, t_Wv, 1024, 1024, 1024);
  convt(syn_w, t_syn, 2048, 1536, 2048);
  convt(out_w, t_out, COUT, 1024, COUTP);

  // ---- phase 1: static precompute ----
  // kv = LN(x @ kv_w.T + kv_b)
  gemm(x, CFEAT, t_kv, 512, kv_b, 512, kvbuf, CDI, CB * CS, CDI, CFEAT);
  k_ln_rows<<<CB * CS, 256, 0, stream>>>(kvbuf, ln_kv_s, ln_kv_b, CDI);
  // Kc / Vc for cross attention
  gemm(kvbuf, CDI, t_wk_c, 512, in_proj_b + 512, 512, Kcbuf, CDI, CB * CS, CDI, CDI);
  gemm(kvbuf, CDI, t_wv_c, 512, in_proj_b + 1024, 512, Vcbuf, CDI, CB * CS, CDI, CDI);
  // initial state: projections of start_activated_state, broadcast caches
  k_init_proj<<<12, 256, 0, stream>>>(start_act, W_q, W_k, W_v, qkv0);
  k_fill_cache<<<DIV_UP(CB * CMEMP * CDM, 256), 256, 0, stream>>>(qkv0, Qcache, Kcache);
  k_fill_act_v<<<DIV_UP(CB * CDM, 256), 256, 0, stream>>>(qkv0, start_act, vbuf, actbuf);
  k_fill_st<<<DIV_UP(CB * CDM * CMEM, 256), 256, 0, stream>>>(start_tr, stbuf);
  // sync_a for tick 0 = sync_attn(hist0, act0)
  k_sync_attn<<<CB * CSH, 256, 0, stream>>>(Qcache, Kcache, vbuf, syncv);

  // ---- phase 2: 50 thought ticks ----
  for (int t = 0; t < CIT; ++t) {
    const int slot = t % CMEM;
    // cross attention on sync_a
    gemm(syncv, CDM, t_qw, 512, q_b, 512, qbuf, CDI, CB, CDI, CDM);
    gemm(qbuf, CDI, t_wq_c, 512, in_proj_b, 512, qh, CDI, CB, CDI, CDI);
    k_cross_attn<<<CB * CH, 256, 0, stream>>>(qh, Kcbuf, Vcbuf, oatt);
    gemm(oatt, CDI, t_op, 512, out_proj_b, 512, aobuf, CDI, CB, CDI, CDI);
    // synapse
    k_concat<<<DIV_UP(CB * 1536, 256), 256, 0, stream>>>(aobuf, actbuf, pre);
    gemm(pre, 1536, t_syn, 2048, syn_b, 2048, zbuf, 2048, CB, 2048, 1536);
    k_syn_glu_ln<<<CB, 256, 0, stream>>>(zbuf, syn_ln_s, syn_ln_b, stbuf, slot);
    // NLM -> new activations (in place: old act already consumed)
    k_nlm<<<dim3(CDM / 256, CB), 256, 0, stream>>>(stbuf, nlm_w1, nlm_b1,
                                                   nlm_w2, nlm_b2, actbuf, t);
    // project new activation, update circular caches + V
    gemm(actbuf, CDM, t_Wq, 1024, nullptr, 0, qnew, CDM, CB, CDM, CDM);
    gemm(actbuf, CDM, t_Wk, 1024, nullptr, 0, knew, CDM, CB, CDM, CDM);
    gemm(actbuf, CDM, t_Wv, 1024, nullptr, 0, vbuf, CDM, CB, CDM, CDM);
    k_cache_update<<<DIV_UP(CB * CDM, 256), 256, 0, stream>>>(qnew, knew,
                                                              Qcache, Kcache, slot);
    // sync_o (== next tick's sync_a)
    k_sync_attn<<<CB * CSH, 256, 0, stream>>>(Qcache, Kcache, vbuf, syncv);
    // readout
    gemm(syncv, CDM, t_out, COUTP, out_b, COUT, predb, COUTP, CB, COUTP, CDM);
    k_pred_out<<<CB, 256, 0, stream>>>(predb, out, t);
  }

  // syncs[-1]
  k_copy_sync<<<DIV_UP(CB * CDM, 256), 256, 0, stream>>>(syncv, out);
}
